// AlibiUnidirectionalAttention_21182778704504
// MI455X (gfx1250) — matrix operand order repaired and checked by reading, not compiled, not run on hardware
//
#include <hip/hip_runtime.h>

typedef __attribute__((ext_vector_type(16))) __bf16 v16bf;
typedef __attribute__((ext_vector_type(8)))  __bf16 v8bf;
typedef __attribute__((ext_vector_type(8)))  float  v8f;
typedef __attribute__((ext_vector_type(4)))  unsigned int v4ui;
typedef __attribute__((ext_vector_type(8)))  int v8i;
typedef __attribute__((ext_vector_type(4)))  int v4i;

#define BATCH   2
#define SEQ     2048
#define DMODEL  2048
#define NHEADS  16
#define HSIZE   128
#define NQKV    (3 * NHEADS * HSIZE)   /* 6144 */
#define MTOT    (BATCH * SEQ)          /* 4096 */

// ---------- helpers ----------

static __device__ __forceinline__ unsigned short f2bf(float f) {
  unsigned u = __builtin_bit_cast(unsigned, f);
  u += 0x7FFFu + ((u >> 16) & 1u);          // round-to-nearest-even
  return (unsigned short)(u >> 16);
}

// A operand (16x32 bf16, interleaved K): lane L = row (L&15);
// elems[0..7] = p[kbA..kbA+7], elems[8..15] = p[kbA+16..kbA+23]; kbA = (L>=16)*8
static __device__ __forceinline__ v16bf loadA_bf16(const unsigned short* p, int kbA) {
  v8bf lo = *(const v8bf*)(p + kbA);
  v8bf hi = *(const v8bf*)(p + kbA + 16);
  v16bf r;
#pragma unroll
  for (int i = 0; i < 8; ++i) { r[i] = lo[i]; r[i + 8] = hi[i]; }
  return r;
}

// B operand (32x16 bf16, interleaved K): lane L = col (L&15); elems[0..7] = p[kbB/2..kbB/2+7], elems[8..15] = p[kbB/2+16..kbB/2+23]; kbB = (L>=16)*16
static __device__ __forceinline__ v16bf loadB_bf16(const unsigned short* p, int kbB) {
  v8bf lo = *(const v8bf*)(p + (kbB >> 1));
  v8bf hi = *(const v8bf*)(p + (kbB >> 1) + 16);
  v16bf r;
#pragma unroll
  for (int i = 0; i < 8; ++i) { r[i] = lo[i]; r[i + 8] = hi[i]; }
  return r;
}

static __device__ __forceinline__ v8f zero8() {
  v8f z;
#pragma unroll
  for (int i = 0; i < 8; ++i) z[i] = 0.0f;
  return z;
}

// ---------- TDM: 2-D tile load global -> LDS (bf16 elements) ----------
// D# per CDNA5 ISA 8.3/8.4: group0 {count=1, lds_addr, global_addr, type=2},
// group1 {data_size=2B, tensor_dim0/1, tile_dim0/1, tensor_dim0_stride}; groups 2/3 = 0 (2-D).
static __device__ __forceinline__ void tdm_load_2d_bf16(
    const unsigned short* gptr, void* lds_ptr,
    unsigned tensor_d0, unsigned tensor_d1, unsigned stride0,
    unsigned tile_d0, unsigned tile_d1)
{
  unsigned long long ga = (unsigned long long)gptr;
  unsigned lds_off = (unsigned)(unsigned long long)lds_ptr;  // LDS aperture: addr[31:0]
  v4ui g0;
  g0[0] = 1u;                                               // count=1
  g0[1] = lds_off;
  g0[2] = (unsigned)(ga & 0xFFFFFFFFu);
  g0[3] = (unsigned)((ga >> 32) & 0x01FFFFFFu) | (2u << 30);// type=2
  v8i g1;
  g1[0] = (int)(1u << 16);                                  // data_size = 2 bytes
  g1[1] = (int)((tensor_d0 & 0xFFFFu) << 16);
  g1[2] = (int)(((tensor_d0 >> 16) & 0xFFFFu) | ((tensor_d1 & 0xFFFFu) << 16));
  g1[3] = (int)(((tensor_d1 >> 16) & 0xFFFFu) | ((tile_d0 & 0xFFFFu) << 16));
  g1[4] = (int)(tile_d1 & 0xFFFFu);
  g1[5] = (int)stride0;
  g1[6] = 0;
  g1[7] = 0;
  v4i z4; z4[0] = 0; z4[1] = 0; z4[2] = 0; z4[3] = 0;
  v8i z8;
#pragma unroll
  for (int i = 0; i < 8; ++i) z8[i] = 0;
  __builtin_amdgcn_tensor_load_to_lds(g0, g1, z4, z4, z8, 0);
}

// ---------- kernel 1a: elementwise f32 -> bf16 ----------
__global__ void cvt_bf16(const float* __restrict__ in, unsigned short* __restrict__ out,
                         long long n) {
  for (long long i = (long long)blockIdx.x * blockDim.x + threadIdx.x; i < n;
       i += (long long)gridDim.x * blockDim.x)
    out[i] = f2bf(in[i]);
}

// ---------- kernel 1b: transpose + f32 -> bf16 (weights) ----------
__global__ void transpose_cvt_bf16(const float* __restrict__ in,
                                   unsigned short* __restrict__ out, int R, int C) {
  long long total = (long long)R * C;
  for (long long idx = (long long)blockIdx.x * blockDim.x + threadIdx.x; idx < total;
       idx += (long long)gridDim.x * blockDim.x) {
    int r = (int)(idx / C);
    int c = (int)(idx % C);
    out[(long long)c * R + r] = f2bf(in[idx]);
  }
}

// ---------- kernel 2: QKV GEMM, TDM-staged B, bf16 WMMA ----------
__global__ void __launch_bounds__(256) qkv_gemm(const unsigned short* __restrict__ xb,
                                                const unsigned short* __restrict__ wT,
                                                const float* __restrict__ bias,
                                                unsigned short* __restrict__ Q,
                                                unsigned short* __restrict__ K,
                                                unsigned short* __restrict__ Vt)
{
  __shared__ __align__(16) unsigned short bstage[2][64 * 32];
  const int lane = threadIdx.x & 31;
  const int wv   = threadIdx.x >> 5;
  const int nBlocksN = NQKV / 64;                // 96
  const int Mbase = (blockIdx.x / nBlocksN) * 128 + wv * 16;
  const int Nbase = (blockIdx.x % nBlocksN) * 64;
  const int lr  = lane & 15;
  const int kbA = (lane >> 4) << 3;
  const int kbB = (lane >> 4) << 4;

  const unsigned short* arow = xb + (long long)(Mbase + lr) * DMODEL;

  if (wv == 0) {
    tdm_load_2d_bf16(wT + (long long)Nbase * DMODEL, &bstage[0][0],
                     DMODEL, NQKV, DMODEL, 32, 64);
    __builtin_amdgcn_s_wait_tensorcnt(0);
  }
  __syncthreads();

  v8f acc[4];
#pragma unroll
  for (int t = 0; t < 4; ++t) acc[t] = zero8();

  for (int k = 0; k < DMODEL; k += 32) {
    const int buf = (k >> 5) & 1;
    if (wv == 0 && (k + 32) < DMODEL)
      tdm_load_2d_bf16(wT + (long long)Nbase * DMODEL + (k + 32), &bstage[buf ^ 1][0],
                       DMODEL, NQKV, DMODEL, 32, 64);
    // preload all operands into distinct registers, then chain WMMAs
    v16bf a = loadA_bf16(arow + k, kbA);
    v16bf bm[4];
#pragma unroll
    for (int t = 0; t < 4; ++t)
      bm[t] = loadB_bf16(&bstage[buf][(t * 16 + lr) * 32], kbB);
#pragma unroll
    for (int t = 0; t < 4; ++t)
      acc[t] = __builtin_amdgcn_wmma_f32_16x16x32_bf16(false, a, false, bm[t],
                                                       (short)0, acc[t], false, false);
    if (wv == 0) __builtin_amdgcn_s_wait_tensorcnt(0);
    __syncthreads();
  }

  const int rowHi = kbA;
#pragma unroll
  for (int t = 0; t < 4; ++t) {
    int n = Nbase + t * 16 + lr;
    float bv = bias[n];
    int part = n >> 11;                          // 0=q 1=k 2=v
    int c = n & 2047, h = c >> 7, d = c & 127;
#pragma unroll
    for (int r = 0; r < 8; ++r) {
      int m = Mbase + rowHi + r;
      int bb = m >> 11, s = m & 2047;
      int bh = bb * NHEADS + h;
      unsigned short o = f2bf(acc[t][r] + bv);
      if (part == 0)      Q[((long long)bh * SEQ + s) * HSIZE + d] = o;
      else if (part == 1) K[((long long)bh * SEQ + s) * HSIZE + d] = o;
      else                Vt[((long long)bh * HSIZE + d) * SEQ + s] = o;
    }
  }
}

// ---------- kernel 3: flash attention with ALiBi + causal mask ----------
__global__ void __launch_bounds__(32) attn_kernel(const unsigned short* __restrict__ Q,
                                                  const unsigned short* __restrict__ K,
                                                  const unsigned short* __restrict__ Vt,
                                                  unsigned short* __restrict__ ctx)
{
  __shared__ __align__(16) unsigned short pbuf[16 * 32];
  const int lane  = threadIdx.x & 31;
  const int bh    = blockIdx.x >> 7;
  const int itile = blockIdx.x & 127;
  const int ibase = itile * 16;
  const int b = bh >> 4, h = bh & 15;
  const float slope = exp2f(-0.5f * (float)h);   // 1 / 2^(h/2)

  const int lr  = lane & 15;
  const int kbA = (lane >> 4) << 3;
  const int kbB = (lane >> 4) << 4;
  const int rowHi = kbA;

  const unsigned short* Qb = Q  + (long long)bh * SEQ * HSIZE;
  const unsigned short* Kb = K  + (long long)bh * SEQ * HSIZE;
  const unsigned short* Vb = Vt + (long long)bh * HSIZE * SEQ;

  v16bf qa[4];
#pragma unroll
  for (int t = 0; t < 4; ++t)
    qa[t] = loadA_bf16(Qb + (long long)(ibase + lr) * HSIZE + t * 32, kbA);

  v8f cacc[8];
#pragma unroll
  for (int d = 0; d < 8; ++d) cacc[d] = zero8();
  float mrow[8], lrow[8];
#pragma unroll
  for (int r = 0; r < 8; ++r) { mrow[r] = -__builtin_inff(); lrow[r] = 0.0f; }

  const int jend = ibase + 16;
  for (int jb = 0; jb < jend; jb += 32) {
    // preload all 8 K B-tiles, then chain the 8 score WMMAs
    v16bf kb0[4], kb1[4];
#pragma unroll
    for (int t = 0; t < 4; ++t)
      kb0[t] = loadB_bf16(Kb + (long long)(jb + lr) * HSIZE + t * 32, kbB);
#pragma unroll
    for (int t = 0; t < 4; ++t)
      kb1[t] = loadB_bf16(Kb + (long long)(jb + 16 + lr) * HSIZE + t * 32, kbB);

    v8f s0 = zero8(), s1 = zero8();
#pragma unroll
    for (int t = 0; t < 4; ++t)
      s0 = __builtin_amdgcn_wmma_f32_16x16x32_bf16(false, qa[t], false, kb0[t],
                                                   (short)0, s0, false, false);
#pragma unroll
    for (int t = 0; t < 4; ++t)
      s1 = __builtin_amdgcn_wmma_f32_16x16x32_bf16(false, qa[t], false, kb1[t],
                                                   (short)0, s1, false, false);

    float pm[8];
#pragma unroll
    for (int r = 0; r < 8; ++r) {
      int i  = ibase + rowHi + r;
      int j0 = jb + lr;
      int j1 = jb + 16 + lr;
      float v0 = (j0 <= i) ? (s0[r] + slope * (float)(j0 - i)) : -__builtin_inff();
      float v1 = (j1 <= i) ? (s1[r] + slope * (float)(j1 - i)) : -__builtin_inff();
      s0[r] = v0; s1[r] = v1;
      float mx = fmaxf(v0, v1);
      mx = fmaxf(mx, __shfl_xor(mx, 1));
      mx = fmaxf(mx, __shfl_xor(mx, 2));
      mx = fmaxf(mx, __shfl_xor(mx, 4));
      mx = fmaxf(mx, __shfl_xor(mx, 8));
      pm[r] = mx;
    }

    float scl[8];
#pragma unroll
    for (int r = 0; r < 8; ++r) {
      float mnew = fmaxf(mrow[r], pm[r]);
      float sc = __expf(mrow[r] - mnew);
      float p0 = __expf(s0[r] - mnew);
      float p1 = __expf(s1[r] - mnew);
      float rs = p0 + p1;
      rs += __shfl_xor(rs, 1);
      rs += __shfl_xor(rs, 2);
      rs += __shfl_xor(rs, 4);
      rs += __shfl_xor(rs, 8);
      lrow[r] = lrow[r] * sc + rs;
      mrow[r] = mnew;
      scl[r]  = sc;
      int row = rowHi + r;
      pbuf[row * 32 + lr]      = f2bf(p0);
      pbuf[row * 32 + 16 + lr] = f2bf(p1);
    }

#pragma unroll
    for (int d = 0; d < 8; ++d)
#pragma unroll
      for (int r = 0; r < 8; ++r)
        cacc[d][r] *= scl[r];

    __asm__ volatile("s_wait_dscnt 0" ::: "memory");

    // PV: preload V B-tiles in groups of 4, chain WMMAs
    v16bf pa = loadA_bf16(&pbuf[lr * 32], kbA);
#pragma unroll
    for (int dd = 0; dd < 8; dd += 4) {
      v16bf vb[4];
#pragma unroll
      for (int t = 0; t < 4; ++t)
        vb[t] = loadB_bf16(Vb + (long long)((dd + t) * 16 + lr) * SEQ + jb, kbB);
#pragma unroll
      for (int t = 0; t < 4; ++t)
        cacc[dd + t] = __builtin_amdgcn_wmma_f32_16x16x32_bf16(false, pa, false, vb[t],
                                                               (short)0, cacc[dd + t],
                                                               false, false);
    }
  }

#pragma unroll
  for (int r = 0; r < 8; ++r) {
    float inv = 1.0f / lrow[r];
    int srow = ibase + rowHi + r;
    long long base = ((long long)b * SEQ + srow) * (NHEADS * HSIZE) + (long long)h * HSIZE;
#pragma unroll
    for (int d = 0; d < 8; ++d)
      ctx[base + d * 16 + lr] = f2bf(cacc[d][r] * inv);
  }
}

// ---------- kernel 4: output projection GEMM, TDM-staged B ----------
__global__ void __launch_bounds__(256) out_gemm(const unsigned short* __restrict__ A,
                                                const unsigned short* __restrict__ BT,
                                                const float* __restrict__ bias,
                                                float* __restrict__ out)
{
  __shared__ __align__(16) unsigned short bstage[2][64 * 32];
  const int lane = threadIdx.x & 31;
  const int wv   = threadIdx.x >> 5;
  const int nBlocksN = DMODEL / 64;              // 32
  const int Mbase = (blockIdx.x / nBlocksN) * 128 + wv * 16;
  const int Nbase = (blockIdx.x % nBlocksN) * 64;
  const int lr  = lane & 15;
  const int kbA = (lane >> 4) << 3;
  const int kbB = (lane >> 4) << 4;

  const unsigned short* arow = A + (long long)(Mbase + lr) * DMODEL;

  if (wv == 0) {
    tdm_load_2d_bf16(BT + (long long)Nbase * DMODEL, &bstage[0][0],
                     DMODEL, DMODEL, DMODEL, 32, 64);
    __builtin_amdgcn_s_wait_tensorcnt(0);
  }
  __syncthreads();

  v8f acc[4];
#pragma unroll
  for (int t = 0; t < 4; ++t) acc[t] = zero8();

  for (int k = 0; k < DMODEL; k += 32) {
    const int buf = (k >> 5) & 1;
    if (wv == 0 && (k + 32) < DMODEL)
      tdm_load_2d_bf16(BT + (long long)Nbase * DMODEL + (k + 32), &bstage[buf ^ 1][0],
                       DMODEL, DMODEL, DMODEL, 32, 64);
    v16bf a = loadA_bf16(arow + k, kbA);
    v16bf bm[4];
#pragma unroll
    for (int t = 0; t < 4; ++t)
      bm[t] = loadB_bf16(&bstage[buf][(t * 16 + lr) * 32], kbB);
#pragma unroll
    for (int t = 0; t < 4; ++t)
      acc[t] = __builtin_amdgcn_wmma_f32_16x16x32_bf16(false, a, false, bm[t],
                                                       (short)0, acc[t], false, false);
    if (wv == 0) __builtin_amdgcn_s_wait_tensorcnt(0);
    __syncthreads();
  }

  const int rowHi = kbA;
#pragma unroll
  for (int t = 0; t < 4; ++t) {
    int n = Nbase + t * 16 + lr;
    float bv = bias[n];
#pragma unroll
    for (int r = 0; r < 8; ++r) {
      int m = Mbase + rowHi + r;
      out[(long long)m * DMODEL + n] = acc[t][r] + bv;
    }
  }
}

// ---------- launch ----------
extern "C" void kernel_launch(void* const* d_in, const int* in_sizes, int n_in,
                              void* d_out, int out_size, void* d_ws, size_t ws_size,
                              hipStream_t stream) {
  (void)in_sizes; (void)n_in; (void)out_size; (void)ws_size;
  const float* x      = (const float*)d_in[0];
  const float* qkv_w  = (const float*)d_in[1];
  const float* qkv_b  = (const float*)d_in[2];
  const float* out_w  = (const float*)d_in[3];
  const float* out_b  = (const float*)d_in[4];
  float* out = (float*)d_out;

  char* p = (char*)d_ws;
  unsigned short* xb  = (unsigned short*)p; p += (size_t)MTOT * DMODEL * 2;
  unsigned short* wT  = (unsigned short*)p; p += (size_t)NQKV * DMODEL * 2;
  unsigned short* owT = (unsigned short*)p; p += (size_t)DMODEL * DMODEL * 2;
  unsigned short* Q   = (unsigned short*)p; p += (size_t)BATCH * NHEADS * SEQ * HSIZE * 2;
  unsigned short* K   = (unsigned short*)p; p += (size_t)BATCH * NHEADS * SEQ * HSIZE * 2;
  unsigned short* Vt  = (unsigned short*)p; p += (size_t)BATCH * NHEADS * HSIZE * SEQ * 2;
  unsigned short* ctx = (unsigned short*)p; p += (size_t)MTOT * DMODEL * 2;

  cvt_bf16<<<4096, 256, 0, stream>>>(x, xb, (long long)MTOT * DMODEL);
  transpose_cvt_bf16<<<4096, 256, 0, stream>>>(qkv_w, wT, DMODEL, NQKV);
  transpose_cvt_bf16<<<2048, 256, 0, stream>>>(out_w, owT, DMODEL, DMODEL);

  qkv_gemm<<<3072, 256, 0, stream>>>(xb, wT, qkv_b, Q, K, Vt);
  attn_kernel<<<BATCH * NHEADS * (SEQ / 16), 32, 0, stream>>>(Q, K, Vt, ctx);
  out_gemm<<<1024, 256, 0, stream>>>(ctx, owT, out_b, out);
}
